// Transformer_MPNN_18279380812413
// MI455X (gfx1250) — matrix operand order repaired and checked by reading, not compiled, not run on hardware
//
#include <hip/hip_runtime.h>
#include <math.h>

typedef __attribute__((ext_vector_type(16))) _Float16 v16h;
typedef __attribute__((ext_vector_type(8)))  float    v8f;

#define HIDDIM 128

// ---------------------------------------------------------------------
// CDNA5 async global->LDS staging (guarded; falls back to float4 copies)
// builtin signature (from hipcc diagnostic): (v4i as1*, v4i as3*, Ii, Ii)
// ---------------------------------------------------------------------
#if __has_builtin(__builtin_amdgcn_global_load_async_to_lds_b128) && \
    __has_builtin(__builtin_amdgcn_s_wait_asynccnt)
#define USE_ASYNC_LDS 1
typedef int v4i_t __attribute__((ext_vector_type(4)));
typedef __attribute__((address_space(1))) v4i_t as1_v4i;
typedef __attribute__((address_space(3))) v4i_t as3_v4i;
__device__ __forceinline__ void async_copy16(float* lds_dst, const float* gsrc) {
  __builtin_amdgcn_global_load_async_to_lds_b128(
      (as1_v4i*)(uintptr_t)gsrc,
      (as3_v4i*)(uintptr_t)lds_dst, 0, 0);
}
__device__ __forceinline__ void async_wait() {
  __builtin_amdgcn_s_wait_asynccnt(0);
}
#else
#define USE_ASYNC_LDS 0
__device__ __forceinline__ void async_copy16(float* lds_dst, const float* gsrc) {
  *(float4*)lds_dst = *(const float4*)gsrc;
}
__device__ __forceinline__ void async_wait() {}
#endif

__device__ __forceinline__ int a_klocal(int i, int koff) {
  // contiguous k order used here (k = 16*(lane/16) + i, same as wswz_kernel): half i of lane -> k offset within 32-wide slab
  return 2 * koff + i;
}

// =====================================================================
// Weight pre-swizzle: W[K,M] f32 -> f16 fragments laid out so that each
// wave/lane reads its 16-half B fragment as one contiguous 32B vector.
// dst[((tile*(K/32)+kk32)*32 + lane)*16 + i] where
//   k = kk32*32 + (lane>>4)*16 + i ; n = tile*16 + (lane&15)
// =====================================================================
__global__ void wswz_kernel(const float* __restrict__ W,
                            _Float16* __restrict__ Wh, int K, int M)
{
  int idx = blockIdx.x * blockDim.x + threadIdx.x;
  if (idx >= K * M) return;
  int k = idx / M, n = idx - k * M;
  int tile = n >> 4, nl = n & 15;
  int kk32 = k >> 5, kr = k & 31;
  int lane = ((kr >> 4) << 4) | nl;
  int half = kr & 15;
  size_t dst = ((size_t)(tile * (K >> 5) + kk32) * 32 + lane) * 16 + half;
  Wh[dst] = (_Float16)W[idx];
}

// =====================================================================
// Single WMMA linear: OUT[nrows, M] = A[nrows,128] @ W[128,M] + bias
// block = 32 * (M/16) threads; one wave per 16-wide column tile.
// =====================================================================
__global__ __launch_bounds__(512) void linear_wmma_kernel(
    const float* __restrict__ A, const _Float16* __restrict__ Wh,
    const float* __restrict__ bias, float* __restrict__ OUT,
    int nrows, int M)
{
  __shared__ float Af[16 * HIDDIM];
  const int K = HIDDIM;
  int row0 = blockIdx.x * 16;
  bool full = (row0 + 16 <= nrows);
  if (full) {
    for (int ci = threadIdx.x; ci < 16 * (K / 4); ci += blockDim.x) {
      int r = ci / (K / 4), c = (ci - r * (K / 4)) * 4;
      async_copy16(&Af[r * K + c], A + (size_t)(row0 + r) * K + c);
    }
    async_wait();
  } else {
    for (int ci = threadIdx.x; ci < 16 * (K / 4); ci += blockDim.x) {
      int r = ci / (K / 4), c = (ci - r * (K / 4)) * 4;
      int rr = row0 + r;
      float4 v = make_float4(0.f, 0.f, 0.f, 0.f);
      if (rr < nrows) v = *(const float4*)(A + (size_t)rr * K + c);
      *(float4*)&Af[r * K + c] = v;
    }
  }
  __syncthreads();
  int wave = threadIdx.x >> 5, lane = threadIdx.x & 31;
  int mloc = lane & 15, koff = (lane >> 4) * 8;
  int ncol = wave * 16 + (lane & 15);
  const _Float16* wp = Wh + ((size_t)(wave * (K >> 5)) * 32 + lane) * 16;
  v8f acc = {};
  for (int kk = 0; kk < K; kk += 32) {
    v16h a, b;
#pragma unroll
    for (int i = 0; i < 16; ++i)
      a[i] = (_Float16)Af[mloc * K + kk + a_klocal(i, koff)];
    b = *(const v16h*)wp;
    wp += 512;                                  // next 32-wide k slab
    acc = __builtin_amdgcn_wmma_f32_16x16x32_f16(false, a, false, b, (short)0,
                                                 acc, false, false);
  }
  float bv = bias[ncol];
  if (full) {
    size_t base = (size_t)(row0 + koff) * M + ncol;
#pragma unroll
    for (int r = 0; r < 8; ++r) { OUT[base] = acc[r] + bv; base += M; }
  } else {
#pragma unroll
    for (int r = 0; r < 8; ++r) {
      int rr = row0 + koff + r;
      if (rr < nrows) OUT[(size_t)rr * M + ncol] = acc[r] + bv;
    }
  }
}

// =====================================================================
// Fused Linear->ReLU->Linear MLP with WMMA, hidden=128, out=128.
// Input row r:
//   base1 == nullptr : dense  base0[r*K1 + c]               (K1 = 32)
//   base1 != nullptr : concat base0[i0*128+c] | base1[i1*128+c-128] (K1=256)
//     i0 = idx0 ? idx0[r] : r ;  i1 = idx1 ? idx1[r] : r
// OUT[r,128] (+= RES[r,128] if RES)
// =====================================================================
__global__ __launch_bounds__(256) void mlp_wmma_kernel(
    const float* __restrict__ base0, const int* __restrict__ idx0,
    const float* __restrict__ base1, const int* __restrict__ idx1,
    int K1,
    const _Float16* __restrict__ Wh1, const float* __restrict__ b1,
    const _Float16* __restrict__ Wh2, const float* __restrict__ b2,
    float* __restrict__ OUT, const float* __restrict__ RES, int nrows)
{
  __shared__ float    Af[16 * 256];
  __shared__ _Float16 Hh[16 * HIDDIM];
  int row0 = blockIdx.x * 16;
  bool full = (row0 + 16 <= nrows);
  int cpr = K1 >> 2;                            // 16B chunks per row
  for (int ci = threadIdx.x; ci < 16 * cpr; ci += blockDim.x) {
    int r = ci / cpr, c = (ci - r * cpr) * 4;
    int rr = row0 + r;
    const float* g;
    if (base1 == nullptr) {
      g = base0 + (size_t)(rr < nrows ? rr : 0) * K1 + c;
    } else if (c < HIDDIM) {
      int s = (rr < nrows) ? (idx0 ? idx0[rr] : rr) : 0;
      g = base0 + (size_t)s * HIDDIM + c;
    } else {
      int s = (rr < nrows) ? (idx1 ? idx1[rr] : rr) : 0;
      g = base1 + (size_t)s * HIDDIM + (c - HIDDIM);
    }
    if (full) {
      async_copy16(&Af[r * K1 + c], g);
    } else {
      float4 v = make_float4(0.f, 0.f, 0.f, 0.f);
      if (rr < nrows) v = *(const float4*)g;
      *(float4*)&Af[r * K1 + c] = v;
    }
  }
  if (full) async_wait();
  __syncthreads();

  int lane = threadIdx.x & 31, wave = threadIdx.x >> 5;
  int mloc = lane & 15, koff = (lane >> 4) * 8;
  int ncol = wave * 16 + (lane & 15);

  // layer 1: hidden = relu(A @ W1 + b1), W1:[K1,128]
  {
    const _Float16* wp = Wh1 + ((size_t)(wave * (K1 >> 5)) * 32 + lane) * 16;
    v8f acc = {};
    for (int kk = 0; kk < K1; kk += 32) {
      v16h a, b;
#pragma unroll
      for (int i = 0; i < 16; ++i)
        a[i] = (_Float16)Af[mloc * K1 + kk + a_klocal(i, koff)];
      b = *(const v16h*)wp;
      wp += 512;
      acc = __builtin_amdgcn_wmma_f32_16x16x32_f16(false, a, false, b, (short)0,
                                                   acc, false, false);
    }
    float bv = b1[ncol];
#pragma unroll
    for (int r = 0; r < 8; ++r)
      Hh[(koff + r) * HIDDIM + ncol] = (_Float16)fmaxf(acc[r] + bv, 0.0f);
  }
  __syncthreads();

  // layer 2: OUT = Hh @ W2 + b2 (+RES)
  {
    const _Float16* wp = Wh2 + ((size_t)(wave * (HIDDIM >> 5)) * 32 + lane) * 16;
    v8f acc = {};
    for (int kk = 0; kk < HIDDIM; kk += 32) {
      v16h a, b;
#pragma unroll
      for (int i = 0; i < 16; ++i)
        a[i] = Hh[mloc * HIDDIM + kk + a_klocal(i, koff)];
      b = *(const v16h*)wp;
      wp += 512;
      acc = __builtin_amdgcn_wmma_f32_16x16x32_f16(false, a, false, b, (short)0,
                                                   acc, false, false);
    }
    float bv = b2[ncol];
    if (full) {
      size_t base = (size_t)(row0 + koff) * HIDDIM + ncol;
#pragma unroll
      for (int r = 0; r < 8; ++r) {
        float v = acc[r] + bv;
        if (RES) v += RES[base];
        OUT[base] = v;
        base += HIDDIM;
      }
    } else {
#pragma unroll
      for (int r = 0; r < 8; ++r) {
        int rr = row0 + koff + r;
        if (rr < nrows) {
          float v = acc[r] + bv;
          if (RES) v += RES[(size_t)rr * HIDDIM + ncol];
          OUT[(size_t)rr * HIDDIM + ncol] = v;
        }
      }
    }
  }
}

// =====================================================================
// MPNN aggregation helpers
// =====================================================================
__global__ void count_edges_kernel(const int* __restrict__ dst,
                                   float* __restrict__ cnt, int E)
{
  int i = blockIdx.x * blockDim.x + threadIdx.x;
  if (i < E) atomicAdd(&cnt[dst[i]], 1.0f);
}

__global__ void scatter_add_kernel(const float* __restrict__ M,
                                   const int* __restrict__ dst,
                                   float* __restrict__ agg, int E)
{
  int idx = blockIdx.x * blockDim.x + threadIdx.x;   // E*32 threads, 4 floats each
  if (idx >= E * 32) return;
  int e = idx >> 5, c = (idx & 31) * 4;
  int d = dst[e];
  const float* mp = M + (size_t)e * HIDDIM + c;
  float* ap = agg + (size_t)d * HIDDIM + c;
#pragma unroll
  for (int j = 0; j < 4; ++j) atomicAdd(ap + j, mp[j]);
}

__global__ void div_cnt_kernel(float* __restrict__ agg,
                               const float* __restrict__ cnt, int n)
{
  int idx = blockIdx.x * blockDim.x + threadIdx.x;   // N*128
  if (idx >= n) return;
  agg[idx] /= fmaxf(cnt[idx >> 7], 1.0f);
}

// =====================================================================
// TransformerConv attention kernels (HEADS=2, HID=128)
// =====================================================================
__global__ __launch_bounds__(256) void attn_alpha_kernel(
    const float* __restrict__ Q, const float* __restrict__ Km,
    const int* __restrict__ src, const int* __restrict__ dst,
    float* __restrict__ alpha, unsigned* __restrict__ amaxu, int E)
{
  int e = blockIdx.x * (blockDim.x >> 5) + (threadIdx.x >> 5);
  int lane = threadIdx.x & 31;
  if (e >= E) return;
  int s = src[e], d = dst[e];
  const float* qd = Q + (size_t)d * 256;
  const float* ks = Km + (size_t)s * 256;
#pragma unroll
  for (int h = 0; h < 2; ++h) {
    int base = h * HIDDIM + lane * 4;
    float p = qd[base + 0] * ks[base + 0] + qd[base + 1] * ks[base + 1] +
              qd[base + 2] * ks[base + 2] + qd[base + 3] * ks[base + 3];
#pragma unroll
    for (int off = 16; off > 0; off >>= 1) p += __shfl_xor(p, off, 32);
    if (lane == 0) {
      float a = p * 0.08838834764831845f;        // 1/sqrt(128)
      alpha[(size_t)e * 2 + h] = a;
      unsigned ub = __float_as_uint(a);
      unsigned m = (ub & 0x80000000u) ? ~ub : (ub | 0x80000000u); // monotonic map
      atomicMax(&amaxu[(size_t)d * 2 + h], m);
    }
  }
}

__global__ void attn_exp_kernel(float* __restrict__ alpha,
                                const unsigned* __restrict__ amaxu,
                                float* __restrict__ den,
                                const int* __restrict__ dst, int E)
{
  int i = blockIdx.x * blockDim.x + threadIdx.x;   // E*2
  if (i >= E * 2) return;
  int e = i >> 1, h = i & 1;
  int d = dst[e];
  unsigned u = amaxu[(size_t)d * 2 + h];
  float amax = __uint_as_float((u & 0x80000000u) ? (u & 0x7FFFFFFFu) : ~u);
  float ex = expf(alpha[i] - amax);
  alpha[i] = ex;
  atomicAdd(&den[(size_t)d * 2 + h], ex);
}

__global__ void attn_scatter_kernel(const float* __restrict__ V,
                                    const float* __restrict__ alpha,
                                    const float* __restrict__ den,
                                    const int* __restrict__ src,
                                    const int* __restrict__ dst,
                                    float* __restrict__ out2, int E)
{
  int idx = blockIdx.x * blockDim.x + threadIdx.x;  // E*64 threads, 4 floats each
  if (idx >= E * 64) return;
  int e = idx >> 6, g = (idx & 63) * 4;             // g in [0,256)
  int h = g >> 7;
  int s = src[e], d = dst[e];
  float w = alpha[(size_t)e * 2 + h] / (den[(size_t)d * 2 + h] + 1e-16f);
  const float* vp = V + (size_t)s * 256 + g;
  float* op = out2 + (size_t)d * 256 + g;
#pragma unroll
  for (int j = 0; j < 4; ++j) atomicAdd(op + j, w * vp[j]);
}

// h = LN( mean_heads(out2) + skip + h ) * g + b   (wave per node)
__global__ __launch_bounds__(256) void tconv_ln_kernel(
    float* __restrict__ H, const float* __restrict__ out2,
    const float* __restrict__ skip, const float* __restrict__ gg,
    const float* __restrict__ bb, int n_nodes)
{
  int node = blockIdx.x * (blockDim.x >> 5) + (threadIdx.x >> 5);
  int lane = threadIdx.x & 31;
  if (node >= n_nodes) return;
  float x[4];
  float sum = 0.0f;
#pragma unroll
  for (int j = 0; j < 4; ++j) {
    int c = lane * 4 + j;
    float v = 0.5f * (out2[(size_t)node * 256 + c] +
                      out2[(size_t)node * 256 + 128 + c]) +
              skip[(size_t)node * HIDDIM + c] + H[(size_t)node * HIDDIM + c];
    x[j] = v;
    sum += v;
  }
#pragma unroll
  for (int off = 16; off > 0; off >>= 1) sum += __shfl_xor(sum, off, 32);
  float mu = sum * (1.0f / 128.0f);
  float vs = 0.0f;
#pragma unroll
  for (int j = 0; j < 4; ++j) { float d = x[j] - mu; vs += d * d; }
#pragma unroll
  for (int off = 16; off > 0; off >>= 1) vs += __shfl_xor(vs, off, 32);
  float inv = rsqrtf(vs * (1.0f / 128.0f) + 1e-5f);
#pragma unroll
  for (int j = 0; j < 4; ++j) {
    int c = lane * 4 + j;
    H[(size_t)node * HIDDIM + c] = gg[c] * (x[j] - mu) * inv + bb[c];
  }
}

// =====================================================================
// Pooling + head
// =====================================================================
__global__ void pool_kernel(const float* __restrict__ H,
                            const int* __restrict__ batch,
                            float* __restrict__ pool, int n_nodes)
{
  int idx = blockIdx.x * blockDim.x + threadIdx.x;  // N*32 threads, 4 floats each
  if (idx >= n_nodes * 32) return;
  int n = idx >> 5, c = (idx & 31) * 4;
  int g = batch[n];
  const float* hp = H + (size_t)n * HIDDIM + c;
  float* pp = pool + (size_t)g * HIDDIM + c;
#pragma unroll
  for (int j = 0; j < 4; ++j) atomicAdd(pp + j, hp[j]);
}

__global__ __launch_bounds__(128) void head_kernel(
    const float* __restrict__ pool, const float* __restrict__ W1,
    const float* __restrict__ b1, const float* __restrict__ W2,
    const float* __restrict__ b2, float* __restrict__ out)
{
  __shared__ float hid[HIDDIM];
  int g = blockIdx.x, t = threadIdx.x;
  const float* p = pool + (size_t)g * HIDDIM;
  float acc = b1[t];
  for (int k = 0; k < HIDDIM; ++k) acc += p[k] * W1[k * HIDDIM + t];
  hid[t] = fmaxf(acc, 0.0f) * W2[t];
  __syncthreads();
  for (int s = 64; s > 0; s >>= 1) {
    if (t < s) hid[t] += hid[t + s];
    __syncthreads();
  }
  if (t == 0) out[g] = hid[0] + b2[0];
}

// =====================================================================
// Host orchestration
// =====================================================================
struct LinP { const float* W; const float* b; };
struct ConvP { LinP q, k, v, skip; };
struct LNP { const float* g; const float* b; };

extern "C" void kernel_launch(void* const* d_in, const int* in_sizes, int n_in,
                              void* d_out, int out_size, void* d_ws, size_t ws_size,
                              hipStream_t stream)
{
  const int F_IN = 32;
  const float* X = (const float*)d_in[0];
  const int N = in_sizes[0] / F_IN;       // 60000
  const int G = out_size;                 // 100

  int pi = 1;
  auto lin = [&]() { LinP L{(const float*)d_in[pi], (const float*)d_in[pi + 1]}; pi += 2; return L; };
  auto ln  = [&]() { LNP L{(const float*)d_in[pi], (const float*)d_in[pi + 1]}; pi += 2; return L; };
  auto conv = [&]() { ConvP c; c.q = lin(); c.k = lin(); c.v = lin(); c.skip = lin(); return c; };

  // params in setup_inputs() insertion order
  LinP emb1 = lin(), emb2 = lin();
  struct LayerP { LinP m1, m2, u1, u2; ConvP g2s; LNP lnd; ConvP sub; LNP lns; } lay[2];
  for (int l = 0; l < 2; ++l) {
    lay[l].m1 = lin(); lay[l].m2 = lin();       // mpnn.msg l1,l2
    lay[l].u1 = lin(); lay[l].u2 = lin();       // mpnn.upd l1,l2
    lay[l].g2s = conv(); lay[l].lnd = ln();
    lay[l].sub = conv(); lay[l].lns = ln();
  }
  ConvP s2g = conv(); LNP lna = ln();           // ascend[0]
  LinP head1 = lin(), head2 = lin();

  const int* EI  = (const int*)d_in[pi]; const int EG  = in_sizes[pi] / 2; pi++;
  const int* NSI = (const int*)d_in[pi]; const int ENS = in_sizes[pi] / 2; pi++;
  const int* SEI = (const int*)d_in[pi]; const int ES  = in_sizes[pi] / 2; pi++;
  const int* SNI = (const int*)d_in[pi]; const int ESN = in_sizes[pi] / 2; pi++;
  const int* BATCH = (const int*)d_in[pi]; pi++;

  int Emax = EG; if (ES > Emax) Emax = ES; if (ENS > Emax) Emax = ENS; if (ESN > Emax) Emax = ESN;

  // workspace carve-out (floats)
  float* ws = (float*)d_ws;
  size_t off = 0;
  auto carve = [&](size_t n) { float* p = ws + off; off += (n + 63) & ~(size_t)63; return p; };
  size_t m0sz = (size_t)Emax * HIDDIM;
  if ((size_t)N * 256 > m0sz) m0sz = (size_t)N * 256;
  float* H    = carve((size_t)N * HIDDIM);   // node features
  float* M0   = carve(m0sz);                 // msg output / Q
  float* AGG  = carve((size_t)N * HIDDIM);   // mean-agg / skip
  float* KB   = carve((size_t)N * 256);      // K
  float* VB   = carve((size_t)N * 256);      // V
  float* AT2  = carve((size_t)N * 256);      // attention accumulator [N,2,128]
  float* ALPHA = carve((size_t)Emax * 2);
  float* DEN  = carve((size_t)N * 2);
  unsigned* AMAX = (unsigned*)carve((size_t)N * 2);
  float* CNT  = carve((size_t)N);
  float* POOL = carve((size_t)G * HIDDIM);
  _Float16* HARENA = (_Float16*)carve(1 << 20);   // 1M floats = 2M halves arena
  (void)ws_size; (void)n_in;

  // ---- pre-convert + pre-swizzle all WMMA weights to f16 fragments
  size_t hoff = 0;
  auto wconv = [&](const float* W, int K, int M) -> const _Float16* {
    _Float16* p = HARENA + hoff;
    hoff += (size_t)K * M;
    wswz_kernel<<<(K * M + 255) / 256, 256, 0, stream>>>(W, p, K, M);
    return p;
  };
  struct ConvH { const _Float16 *q, *k, *v, *skip; };
  auto convh = [&](const ConvP& c) {
    ConvH h;
    h.q = wconv(c.q.W, 128, 256);
    h.k = wconv(c.k.W, 128, 256);
    h.v = wconv(c.v.W, 128, 256);
    h.skip = wconv(c.skip.W, 128, 128);
    return h;
  };
  const _Float16* e1h = wconv(emb1.W, F_IN, 128);
  const _Float16* e2h = wconv(emb2.W, 128, 128);
  struct LayerH { const _Float16 *m1, *m2, *u1, *u2; ConvH g2s, sub; } layh[2];
  for (int l = 0; l < 2; ++l) {
    layh[l].m1 = wconv(lay[l].m1.W, 256, 128);
    layh[l].m2 = wconv(lay[l].m2.W, 128, 128);
    layh[l].u1 = wconv(lay[l].u1.W, 256, 128);
    layh[l].u2 = wconv(lay[l].u2.W, 128, 128);
    layh[l].g2s = convh(lay[l].g2s);
    layh[l].sub = convh(lay[l].sub);
  }
  ConvH s2gh = convh(s2g);

  const int nb16 = (N + 15) / 16;

  // ---- embedder: h = MLP(emb, x), K1=32
  mlp_wmma_kernel<<<nb16, 256, 0, stream>>>(X, nullptr, nullptr, nullptr, F_IN,
      e1h, emb1.b, e2h, emb2.b, H, nullptr, N);

  // edge counts for mpnn mean-aggregation (static across layers)
  (void)hipMemsetAsync(CNT, 0, (size_t)N * sizeof(float), stream);
  count_edges_kernel<<<(EG + 255) / 256, 256, 0, stream>>>(EI + EG, CNT, EG);

  auto tconv = [&](const ConvP& cv, const ConvH& ch, const LNP& lnp,
                   const int* ei, int E) {
    const int* srcp = ei;
    const int* dstp = ei + E;
    linear_wmma_kernel<<<nb16, 512, 0, stream>>>(H, ch.q, cv.q.b, M0, N, 256);
    linear_wmma_kernel<<<nb16, 512, 0, stream>>>(H, ch.k, cv.k.b, KB, N, 256);
    linear_wmma_kernel<<<nb16, 512, 0, stream>>>(H, ch.v, cv.v.b, VB, N, 256);
    linear_wmma_kernel<<<nb16, 256, 0, stream>>>(H, ch.skip, cv.skip.b, AGG, N, 128);
    (void)hipMemsetAsync(AMAX, 0, (size_t)N * 2 * sizeof(unsigned), stream);
    (void)hipMemsetAsync(DEN, 0, (size_t)N * 2 * sizeof(float), stream);
    (void)hipMemsetAsync(AT2, 0, (size_t)N * 256 * sizeof(float), stream);
    attn_alpha_kernel<<<(E + 7) / 8, 256, 0, stream>>>(M0, KB, srcp, dstp, ALPHA, AMAX, E);
    attn_exp_kernel<<<(E * 2 + 255) / 256, 256, 0, stream>>>(ALPHA, AMAX, DEN, dstp, E);
    attn_scatter_kernel<<<(E * 64 + 255) / 256, 256, 0, stream>>>(VB, ALPHA, DEN, srcp, dstp, AT2, E);
    tconv_ln_kernel<<<(N + 7) / 8, 256, 0, stream>>>(H, AT2, AGG, lnp.g, lnp.b, N);
  };

  for (int l = 0; l < 2; ++l) {
    // ---- MPNN: m = MLP(msg, [h[src], h[dst]]) per edge (fused, WMMA)
    mlp_wmma_kernel<<<(EG + 15) / 16, 256, 0, stream>>>(H, EI, H, EI + EG, 256,
        layh[l].m1, lay[l].m1.b, layh[l].m2, lay[l].m2.b, M0, nullptr, EG);
    (void)hipMemsetAsync(AGG, 0, (size_t)N * HIDDIM * sizeof(float), stream);
    scatter_add_kernel<<<(EG * 32 + 255) / 256, 256, 0, stream>>>(M0, EI + EG, AGG, EG);
    div_cnt_kernel<<<(N * HIDDIM + 255) / 256, 256, 0, stream>>>(AGG, CNT, N * HIDDIM);
    // h = MLP(upd, [h, agg]) + h   (residual fused)
    mlp_wmma_kernel<<<nb16, 256, 0, stream>>>(H, nullptr, AGG, nullptr, 256,
        layh[l].u1, lay[l].u1.b, layh[l].u2, lay[l].u2.b, H, H, N);
    // ---- ground->sub tconv + LN, then sub tconv + LN
    tconv(lay[l].g2s, layh[l].g2s, lay[l].lnd, NSI, ENS);
    tconv(lay[l].sub, layh[l].sub, lay[l].lns, SEI, ES);
  }
  // ---- ascend
  tconv(s2g, s2gh, lna, SNI, ESN);

  // ---- pool + head
  (void)hipMemsetAsync(POOL, 0, (size_t)G * HIDDIM * sizeof(float), stream);
  pool_kernel<<<(N * 32 + 255) / 256, 256, 0, stream>>>(H, BATCH, POOL, N);
  head_kernel<<<G, 128, 0, stream>>>(POOL, head1.W, head1.b, head2.W, head2.b,
                                     (float*)d_out);
}
